// TransformerXLForQuestionAnswering_68942815035639
// MI455X (gfx1250) — matrix operand order repaired and checked by reading, not compiled, not run on hardware
//
#include <hip/hip_runtime.h>
#include <hip/hip_bf16.h>

// ---------------------------------------------------------------------------
// Transformer-XL QA forward for MI455X (gfx1250), wave32 + WMMA f16->f32.
// All heavy GEMMs and attention contractions run on v_wmma_f32_16x16x32_f16.
// ---------------------------------------------------------------------------

typedef __attribute__((ext_vector_type(16))) _Float16 v16h;
typedef __attribute__((ext_vector_type(8)))  float    v8f;

union V16H { v16h v; _Float16 h[16]; unsigned int u[8]; };

__device__ inline v8f wmma_f16(v16h a, v16h b, v8f c) {
  return __builtin_amdgcn_wmma_f32_16x16x32_f16(false, a, false, b, (short)0, c,
                                                false, false);
}

// Model constants
#define TQ   1024
#define TB   4
#define TML  512
#define TK   1536          // TML + TQ
#define TD   768
#define TH   12
#define TDH  64
#define TDI  3072
#define TL   12
#define TSCALE 0.125f      // 1/sqrt(64)
#define TEMB_SCALE 27.712812921102035f

// ---------------------------------------------------------------------------
// Generic WMMA GEMM: C[M,N] = A[M,K] * B[K,N] (+bias)(+relu)
// A,B row-major f32 (converted to f16 on LDS stage-in), C f32.
// Requires M%64==0, N%64==0, K%32==0.  256 threads (8 waves), 64x64 tile.
// ---------------------------------------------------------------------------
template<bool BIAS, bool RELU>
__global__ __launch_bounds__(256)
void gemm_f32_wmma(const float* __restrict__ A, const float* __restrict__ Bw,
                   const float* __restrict__ bias, float* __restrict__ C,
                   int M, int N, int K) {
  __shared__ _Float16 sA[64][34];
  __shared__ _Float16 sBt[64][34];
  const int tid = threadIdx.x;
  const int wid = tid >> 5;
  const int lane = tid & 31;
  const int lm = lane & 15;
  const int kh = lane >> 4;
  const int m0 = blockIdx.y * 64;
  const int n0 = blockIdx.x * 64;
  const int mBase = (wid & 3) * 16;
  const int nBase = (wid >> 2) * 32;

  v8f acc0 = {};
  v8f acc1 = {};

  for (int k0 = 0; k0 < K; k0 += 32) {
    __syncthreads();
#pragma unroll
    for (int j = 0; j < 8; ++j) {       // A tile 64x32
      int e = tid + j * 256;
      int r = e >> 5, c = e & 31;
      sA[r][c] = (_Float16)A[(size_t)(m0 + r) * K + k0 + c];
    }
#pragma unroll
    for (int j = 0; j < 8; ++j) {       // B tile 32x64 -> transposed [n][k]
      int e = tid + j * 256;
      int r = e >> 6, c = e & 63;
      sBt[c][r] = (_Float16)Bw[(size_t)(k0 + r) * N + n0 + c];
    }
    // prefetch next K-step panels into cache (global_prefetch_b8)
    if (k0 + 32 < K) {
      int e = tid;
      int rA = e >> 5, cA = e & 31;
      __builtin_prefetch(&A[(size_t)(m0 + rA) * K + k0 + 32 + cA], 0, 1);
      int rB = e >> 6, cB = e & 63;
      __builtin_prefetch(&Bw[(size_t)(k0 + 32 + rB) * N + n0 + cB], 0, 1);
    }
    __syncthreads();

    V16H af, bf0, bf1;
#pragma unroll
    for (int p = 0; p < 8; ++p) {
      int kk = kh * 16 + 2 * p;
      af.u[p] = *(const unsigned int*)&sA[mBase + lm][kk];
    }
#pragma unroll
    for (int p = 0; p < 8; ++p) {
      int cc = kh * 16 + 2 * p;
      bf0.u[p] = *(const unsigned int*)&sBt[nBase + lm][cc];
      bf1.u[p] = *(const unsigned int*)&sBt[nBase + 16 + lm][cc];
    }
    acc0 = wmma_f16(af.v, bf0.v, acc0);
    acc1 = wmma_f16(af.v, bf1.v, acc1);
  }

#pragma unroll
  for (int e = 0; e < 8; ++e) {
    int m = m0 + mBase + e + kh * 8;
    int na = n0 + nBase + lm;
    int nb = na + 16;
    float va = acc0[e], vb = acc1[e];
    if (BIAS) { va += bias[na]; vb += bias[nb]; }
    if (RELU) { va = fmaxf(va, 0.0f); vb = fmaxf(vb, 0.0f); }
    C[(size_t)m * N + na] = va;
    C[(size_t)m * N + nb] = vb;
  }
}

// ---------------------------------------------------------------------------
// Fused attention per (b, h, 16-query tile).
// q layouts: [Q][B][H][DH] f16, k/v: [K][B][H][DH] f16, r: [K][H][DH] f16.
// Scores (AC + rel-shifted BD, masked, scaled) kept as f16 in LDS,
// full-row softmax, then P*V with WMMA.  128 threads (4 waves).
// ---------------------------------------------------------------------------
__global__ __launch_bounds__(128)
void attn_kernel(const _Float16* __restrict__ qrwp,
                 const _Float16* __restrict__ qrrp,
                 const _Float16* __restrict__ kp,
                 const _Float16* __restrict__ vp,
                 const _Float16* __restrict__ rp,
                 float* __restrict__ outp) {
  __shared__ _Float16 S[16][TK];
  __shared__ float G[4][16][36];
  __shared__ float red[16][8];
  __shared__ float rowmax[16];
  __shared__ float rowinv[16];

  const int qt = blockIdx.x;  // 0..63
  const int h  = blockIdx.y;  // 0..11
  const int b  = blockIdx.z;  // 0..3
  const int i0 = qt * 16;
  const int tid = threadIdx.x;
  const int wid = tid >> 5;
  const int lane = tid & 31;
  const int lm = lane & 15;
  const int kh = lane >> 4;

  // ---- Phase 1: scores ----
  for (int jt = wid; jt < TK / 16; jt += 4) {
    const int j0 = jt * 16;
    const bool fullmask = (j0 > i0 + 15 + TML);   // wave-uniform
    v8f accAC = {};
    v8f accG0 = {};
    v8f accG1 = {};
    if (!fullmask) {
      const int tb = j0 - i0 + (TQ - 1) - 15;     // rel-shift window base
#pragma unroll
      for (int cs = 0; cs < TDH; cs += 32) {
        V16H aq, ar, bk, br0, br1;
        const size_t qbase =
            (((size_t)(i0 + lm) * TB + b) * TH + h) * TDH + cs;
#pragma unroll
        for (int p = 0; p < 8; ++p) {
          int ko = kh * 16 + 2 * p;
          aq.u[p] = *(const unsigned int*)(qrwp + qbase + ko);
          ar.u[p] = *(const unsigned int*)(qrrp + qbase + ko);
        }
        const size_t kbase =
            (((size_t)(j0 + lm) * TB + b) * TH + h) * TDH + cs + kh * 16;
        bk.v = *(const v16h*)(kp + kbase);

        int t0 = tb + lm;        t0 = t0 < 0 ? 0 : (t0 > TK - 1 ? TK - 1 : t0);
        int t1 = tb + 16 + lm;   t1 = t1 < 0 ? 0 : (t1 > TK - 1 ? TK - 1 : t1);
        br0.v = *(const v16h*)(rp + ((size_t)t0 * TH + h) * TDH + cs + kh * 16);
        br1.v = *(const v16h*)(rp + ((size_t)t1 * TH + h) * TDH + cs + kh * 16);

        accAC = wmma_f16(aq.v, bk.v, accAC);
        accG0 = wmma_f16(ar.v, br0.v, accG0);
        accG1 = wmma_f16(ar.v, br1.v, accG1);
      }
    }
#pragma unroll
    for (int e = 0; e < 8; ++e) {
      int m = e + kh * 8;
      G[wid][m][lm] = accG0[e];
      G[wid][m][16 + lm] = accG1[e];
    }
    // Branchless assembly: G index lm-m+15 is always within [0,30].
#pragma unroll
    for (int e = 0; e < 8; ++e) {
      int m = e + kh * 8;
      int i = i0 + m;
      int j = j0 + lm;
      float g = G[wid][m][lm - m + 15];
      float val = (accAC[e] + g) * TSCALE;
      bool ok = (!fullmask) && (j <= i + TML);
      S[m][j] = (_Float16)(ok ? val : -60000.0f);
    }
  }
  __syncthreads();

  // ---- Phase 2: softmax stats over each 1536-wide row ----
  {
    int row = tid >> 3, seg = tid & 7;
    float mx = -1e30f;
    for (int j = seg; j < TK; j += 8) mx = fmaxf(mx, (float)S[row][j]);
    red[row][seg] = mx;
  }
  __syncthreads();
  if (tid < 16) {
    float mx = red[tid][0];
#pragma unroll
    for (int s = 1; s < 8; ++s) mx = fmaxf(mx, red[tid][s]);
    rowmax[tid] = mx;
  }
  __syncthreads();
  {
    int row = tid >> 3, seg = tid & 7;
    float mx = rowmax[row];
    float sm = 0.0f;
    for (int j = seg; j < TK; j += 8) sm += __expf((float)S[row][j] - mx);
    red[row][seg] = sm;
  }
  __syncthreads();
  if (tid < 16) {
    float sm = 0.0f;
#pragma unroll
    for (int s = 0; s < 8; ++s) sm += red[tid][s];
    rowinv[tid] = 1.0f / sm;
  }
  __syncthreads();

  // ---- Phase 3: O = P * V (each wave owns one 16-col slice of dh) ----
  v8f accO = {};
  const int nt = wid;  // dh columns nt*16 .. nt*16+15
  for (int kc = 0; kc < TK; kc += 32) {
    V16H ap, bv;
    {
      const int m = lm;
      const float mx = rowmax[m];
      const float inv = rowinv[m];
#pragma unroll
      for (int p = 0; p < 8; ++p) {
        int ko = kh * 16 + 2 * p;
        float p0 = __expf((float)S[m][kc + ko] - mx) * inv;
        float p1 = __expf((float)S[m][kc + ko + 1] - mx) * inv;
        ap.h[2 * p]     = (_Float16)p0;
        ap.h[2 * p + 1] = (_Float16)p1;
      }
    }
    {
      const int n = nt * 16 + lm;
#pragma unroll
      for (int i = 0; i < 16; ++i) {
        int key = kc + kh * 16 + i;
        bv.h[i] = vp[(((size_t)key * TB + b) * TH + h) * TDH + n];
      }
    }
    accO = wmma_f16(ap.v, bv.v, accO);
  }
#pragma unroll
  for (int e = 0; e < 8; ++e) {
    int m = e + kh * 8;
    int i = i0 + m;
    int col = h * TDH + nt * 16 + lm;
    outp[((size_t)i * TB + b) * TD + col] = accO[e];
  }
}

// ---------------------------------------------------------------------------
// Small helper kernels
// ---------------------------------------------------------------------------
__global__ void embed_kernel(const int* __restrict__ ids,
                             const float* __restrict__ emb,
                             float* __restrict__ core) {
  size_t idx = (size_t)blockIdx.x * 256 + threadIdx.x;  // over Q*B*D
  if (idx >= (size_t)TQ * TB * TD) return;
  int d = (int)(idx % TD);
  size_t rb = idx / TD;
  int bb = (int)(rb % TB);
  int i  = (int)(rb / TB);
  int tok = ids[(size_t)bb * TQ + i];
  core[idx] = emb[(size_t)tok * TD + d] * TEMB_SCALE;
}

__global__ void posemb_kernel(float* __restrict__ pe) {
  int idx = blockIdx.x * 256 + threadIdx.x;  // over K*D
  if (idx >= TK * TD) return;
  int k = idx / TD, d = idx % TD;
  float pos = (float)(TK - 1 - k);
  int j = (d < TD / 2) ? d : d - TD / 2;
  float invf = __powf(10000.0f, -(2.0f * (float)j) / (float)TD);
  float x = pos * invf;
  pe[idx] = (d < TD / 2) ? __sinf(x) : __cosf(x);
}

__global__ void concat_kernel(const float* __restrict__ mems_l,
                              const float* __restrict__ core,
                              float* __restrict__ cat) {
  size_t idx = (size_t)blockIdx.x * 256 + threadIdx.x;  // over K*B*D
  const size_t memsz = (size_t)TML * TB * TD;
  if (idx >= (size_t)TK * TB * TD) return;
  cat[idx] = (idx < memsz) ? mems_l[idx] : core[idx - memsz];
}

__global__ void copy_mems_kernel(const float* __restrict__ core,
                                 float* __restrict__ outm) {
  size_t idx = (size_t)blockIdx.x * 256 + threadIdx.x;  // over MLEN*B*D
  if (idx >= (size_t)TML * TB * TD) return;
  outm[idx] = core[idx + (size_t)TML * TB * TD];  // rows 512..1023 of core
}

__global__ void split_qkv(const float* __restrict__ wh,
                          const float* __restrict__ rwb,
                          const float* __restrict__ rrb,
                          _Float16* __restrict__ qrw, _Float16* __restrict__ qrr,
                          _Float16* __restrict__ kd, _Float16* __restrict__ vd) {
  size_t idx = (size_t)blockIdx.x * 256 + threadIdx.x;  // over (K*B)*D
  if (idx >= (size_t)TK * TB * TD) return;
  size_t row = idx / TD;       // kpos*B + b
  int c = (int)(idx % TD);     // h*64 + dh
  const float* wr = wh + row * (3 * TD);
  kd[idx] = (_Float16)wr[TD + c];
  vd[idx] = (_Float16)wr[2 * TD + c];
  int kpos = (int)(row >> 2);
  if (kpos >= TML) {
    size_t qi = idx - (size_t)TML * TB * TD;
    float q = wr[c];
    qrw[qi] = (_Float16)(q + rwb[c]);
    qrr[qi] = (_Float16)(q + rrb[c]);
  }
}

__global__ void tohalf_kernel(const float* __restrict__ src,
                              _Float16* __restrict__ dst, size_t n) {
  size_t idx = (size_t)blockIdx.x * 256 + threadIdx.x;
  if (idx < n) dst[idx] = (_Float16)src[idx];
}

__global__ __launch_bounds__(256)
void ln_kernel(float* __restrict__ core, const float* __restrict__ resid,
               const float* __restrict__ g, const float* __restrict__ bta) {
  __shared__ float red[256];
  const int row = blockIdx.x;
  const int tid = threadIdx.x;
  float x[3];
#pragma unroll
  for (int j = 0; j < 3; ++j) {
    int c = tid + j * 256;
    x[j] = core[(size_t)row * TD + c] + resid[(size_t)row * TD + c];
  }
  red[tid] = x[0] + x[1] + x[2];
  __syncthreads();
  for (int off = 128; off > 0; off >>= 1) {
    if (tid < off) red[tid] += red[tid + off];
    __syncthreads();
  }
  const float mean = red[0] / (float)TD;
  __syncthreads();
  float v = 0.0f;
#pragma unroll
  for (int j = 0; j < 3; ++j) { float d = x[j] - mean; v += d * d; }
  red[tid] = v;
  __syncthreads();
  for (int off = 128; off > 0; off >>= 1) {
    if (tid < off) red[tid] += red[tid + off];
    __syncthreads();
  }
  const float rstd = rsqrtf(red[0] / (float)TD + 1e-5f);
#pragma unroll
  for (int j = 0; j < 3; ++j) {
    int c = tid + j * 256;
    core[(size_t)row * TD + c] = (x[j] - mean) * rstd * g[c] + bta[c];
  }
}

__global__ void qa_kernel(const float* __restrict__ core,
                          const float* __restrict__ qa_w,
                          const float* __restrict__ qa_b,
                          float* __restrict__ out) {
  int idx = blockIdx.x * 256 + threadIdx.x;  // 2 * B * Q
  if (idx >= 2 * TB * TQ) return;
  int c = idx / (TB * TQ);
  int rem = idx % (TB * TQ);
  int bb = rem / TQ;
  int i = rem % TQ;
  const float* rowp = core + ((size_t)i * TB + bb) * TD;
  float acc = qa_b[c];
  for (int d = 0; d < TD; ++d) acc += rowp[d] * qa_w[(size_t)d * 2 + c];
  out[(size_t)c * TB * TQ + (size_t)bb * TQ + i] = acc;
}

// ---------------------------------------------------------------------------
// Host launch
// ---------------------------------------------------------------------------
extern "C" void kernel_launch(void* const* d_in, const int* in_sizes, int n_in,
                              void* d_out, int out_size, void* d_ws, size_t ws_size,
                              hipStream_t stream) {
  (void)in_sizes; (void)n_in; (void)out_size; (void)ws_size;

  const int*   input_ids = (const int*)d_in[0];
  const float* mems      = (const float*)d_in[1];
  const float* emb       = (const float*)d_in[2];
  const float* qkv_w     = (const float*)d_in[3];
  const float* r_w       = (const float*)d_in[4];
  const float* o_w       = (const float*)d_in[5];
  const float* r_w_bias  = (const float*)d_in[6];
  const float* r_r_bias  = (const float*)d_in[7];
  const float* ln1_g     = (const float*)d_in[8];
  const float* ln1_b     = (const float*)d_in[9];
  const float* ff_w1     = (const float*)d_in[10];
  const float* ff_b1     = (const float*)d_in[11];
  const float* ff_w2     = (const float*)d_in[12];
  const float* ff_b2     = (const float*)d_in[13];
  const float* ln2_g     = (const float*)d_in[14];
  const float* ln2_b     = (const float*)d_in[15];
  const float* qa_w      = (const float*)d_in[16];
  const float* qa_b      = (const float*)d_in[17];
  float* out = (float*)d_out;

  // workspace carving (256B aligned)
  char* w = (char*)d_ws;
  auto carve = [&](size_t bytes) -> char* {
    char* p = w;
    w += (bytes + 255) & ~(size_t)255;
    return p;
  };
  const size_t NQB = (size_t)TQ * TB;   // 4096
  const size_t NKB = (size_t)TK * TB;   // 6144

  float*    core  = (float*)   carve(NQB * TD * 4);
  float*    cat   = (float*)   carve(NKB * TD * 4);
  float*    wh    = (float*)   carve(NKB * 3 * TD * 4);
  float*    pe    = (float*)   carve((size_t)TK * TD * 4);
  float*    rb    = (float*)   carve((size_t)TK * TD * 4);
  _Float16* rh    = (_Float16*)carve((size_t)TK * TD * 2);
  _Float16* qrw   = (_Float16*)carve(NQB * TD * 2);
  _Float16* qrr   = (_Float16*)carve(NQB * TD * 2);
  _Float16* kbuf  = (_Float16*)carve(NKB * TD * 2);
  _Float16* vbuf  = (_Float16*)carve(NKB * TD * 2);
  float*    attnb = (float*)   carve(NQB * TD * 4);
  float*    proj  = (float*)   carve(NQB * TD * 4);
  float*    ffh   = (float*)   carve(NQB * TDI * 4);
  float*    ffo   = (float*)   carve(NQB * TD * 4);

  const size_t memLayer = (size_t)TML * TB * TD;  // 1,572,864
  float* out_mems = out + 2 * (size_t)TB * TQ;

  embed_kernel<<<(unsigned)((NQB * TD + 255) / 256), 256, 0, stream>>>(input_ids, emb, core);
  posemb_kernel<<<(TK * TD + 255) / 256, 256, 0, stream>>>(pe);

  for (int l = 0; l < TL; ++l) {
    // new_mems[l] = hids[l][-512:] (core at layer entry)
    copy_mems_kernel<<<(unsigned)((memLayer + 255) / 256), 256, 0, stream>>>(
        core, out_mems + (size_t)l * memLayer);
    concat_kernel<<<(unsigned)((NKB * TD + 255) / 256), 256, 0, stream>>>(
        mems + (size_t)l * memLayer, core, cat);

    // QKV projection: [6144,768] x [768,2304]
    gemm_f32_wmma<false, false><<<dim3(3 * TD / 64, (unsigned)(NKB / 64)), 256, 0, stream>>>(
        cat, qkv_w + (size_t)l * TD * 3 * TD, nullptr, wh, (int)NKB, 3 * TD, TD);
    split_qkv<<<(unsigned)((NKB * TD + 255) / 256), 256, 0, stream>>>(
        wh, r_w_bias + (size_t)l * TD, r_r_bias + (size_t)l * TD, qrw, qrr, kbuf, vbuf);

    // position keys: [1536,768] x [768,768]
    gemm_f32_wmma<false, false><<<dim3(TD / 64, TK / 64), 256, 0, stream>>>(
        pe, r_w + (size_t)l * TD * TD, nullptr, rb, TK, TD, TD);
    tohalf_kernel<<<(TK * TD + 255) / 256, 256, 0, stream>>>(rb, rh, (size_t)TK * TD);

    // fused attention (AC + rel-shift BD + mask + softmax + P*V)
    attn_kernel<<<dim3(TQ / 16, TH, TB), 128, 0, stream>>>(qrw, qrr, kbuf, vbuf, rh, attnb);

    // output projection + LN1
    gemm_f32_wmma<false, false><<<dim3(TD / 64, (unsigned)(NQB / 64)), 256, 0, stream>>>(
        attnb, o_w + (size_t)l * TD * TD, nullptr, proj, (int)NQB, TD, TD);
    ln_kernel<<<(unsigned)NQB, 256, 0, stream>>>(core, proj,
        ln1_g + (size_t)l * TD, ln1_b + (size_t)l * TD);

    // FFN
    gemm_f32_wmma<true, true><<<dim3(TDI / 64, (unsigned)(NQB / 64)), 256, 0, stream>>>(
        core, ff_w1 + (size_t)l * TD * TDI, ff_b1 + (size_t)l * TDI, ffh, (int)NQB, TDI, TD);
    gemm_f32_wmma<true, false><<<dim3(TD / 64, (unsigned)(NQB / 64)), 256, 0, stream>>>(
        ffh, ff_w2 + (size_t)l * TDI * TD, ff_b2 + (size_t)l * TD, ffo, (int)NQB, TD, TDI);
    ln_kernel<<<(unsigned)NQB, 256, 0, stream>>>(core, ffo,
        ln2_g + (size_t)l * TD, ln2_b + (size_t)l * TD);
  }

  qa_kernel<<<(2 * TB * TQ + 255) / 256, 256, 0, stream>>>(core, qa_w, qa_b, out);
}
